// MoERouter_17188459118818
// MI455X (gfx1250) — compile-verified, matrix operand order checked by reading, not run on hardware
//
#include <hip/hip_runtime.h>
#include <math.h>

// ---------------------------------------------------------------------------
// MoE top-1 layer for MI455X (gfx1250, wave32, WMMA, async LDS copies).
// route -> gather tokens per expert (segments padded to 64-row tiles) ->
// one-shot fp32->bf16 weight conversion (bf16 weights ~128MB fit in 192MB L2)
// -> double-buffered bf16 WMMA GEMMs (64x256 tile, 8 accums/wave) with
// global_load_async_to_lds_b128 staging (ASYNCcnt) -> exact GELU -> fc2 +
// gate scatter.
// ---------------------------------------------------------------------------

using bf16 = __bf16;
typedef __attribute__((ext_vector_type(16))) __bf16 v16bf;
typedef __attribute__((ext_vector_type(8)))  float  v8f;

constexpr int T_TOK = 4096;            // BATCH*SEQ
constexpr int HID   = 1024;
constexpr int FFN_  = 4096;
constexpr int NE    = 8;
constexpr int PADR  = 64;              // M-tile / segment alignment
constexpr int TPAD  = T_TOK + NE * PADR;   // 4608
constexpr int MAX_MT = TPAD / PADR;        // 72 fixed M-tiles (graph-safe)

// ---- workspace layout (bytes) ----
constexpr size_t WS_PROBS = 0;                                      // T*8 f32
constexpr size_t WS_GATE  = WS_PROBS + sizeof(float) * T_TOK * 8;
constexpr size_t WS_EIDX  = WS_GATE  + sizeof(float) * T_TOK;
constexpr size_t WS_META  = WS_EIDX  + sizeof(int)   * T_TOK;       // 32 ints
constexpr size_t WS_TILE  = WS_META  + sizeof(int) * 32;            // 128 ints
constexpr size_t WS_ROWTK = WS_TILE  + sizeof(int) * 128;           // TPAD ints
constexpr size_t WS_XG    = ((WS_ROWTK + sizeof(int) * TPAD) + 255) & ~size_t(255);
constexpr size_t WS_H     = WS_XG  + 2ull * TPAD * HID;             // bf16
constexpr size_t WS_W1B   = WS_H   + 2ull * TPAD * FFN_;            // bf16 [E][F][H]
constexpr size_t WS_W2B   = WS_W1B + 2ull * NE * FFN_ * HID;        // bf16 [E][H][F]

// ---------------------------------------------------------------------------
// async copy helpers (gfx1250): 16B global -> LDS per lane, ASYNCcnt tracked.
// ---------------------------------------------------------------------------
__device__ __forceinline__ void async_g2l_b128(uint32_t lds_off, uint32_t goff,
                                               unsigned long long sbase) {
  asm volatile("global_load_async_to_lds_b128 %0, %1, %2"
               :: "v"(lds_off), "v"(goff), "s"(sbase) : "memory");
}
__device__ __forceinline__ void wait_async_le5() {
  asm volatile("s_wait_asynccnt 0x5" ::: "memory");
}
__device__ __forceinline__ void wait_async_0() {
  asm volatile("s_wait_asynccnt 0x0" ::: "memory");
}
__device__ __forceinline__ uint32_t lds_addr32(const void* p) {
  return (uint32_t)(uintptr_t)p;     // LDS generic addr: offset in low 32 bits
}

// ---------------------------------------------------------------------------
// Kernel 0: fp32 -> bf16 bulk conversion (weights), float4 -> 4x bf16.
// ---------------------------------------------------------------------------
__global__ __launch_bounds__(256) void cvt_kernel(
    const float* __restrict__ src, bf16* __restrict__ dst, int n4)
{
  int i = blockIdx.x * 256 + threadIdx.x;
  const int stride = gridDim.x * 256;
  for (; i < n4; i += stride) {
    const float4 w = ((const float4*)src)[i];
    union { bf16 h[4]; uint2 v; } u;
    u.h[0] = (bf16)w.x; u.h[1] = (bf16)w.y; u.h[2] = (bf16)w.z; u.h[3] = (bf16)w.w;
    ((uint2*)dst)[i] = u.v;
  }
}

// ---------------------------------------------------------------------------
// Kernel 1: router. One wave (32 lanes) per token.
// ---------------------------------------------------------------------------
__global__ __launch_bounds__(256) void router_kernel(
    const float* __restrict__ x, const float* __restrict__ rw,
    float* __restrict__ probs, float* __restrict__ gate, int* __restrict__ eidx)
{
  const int wid  = threadIdx.x >> 5;
  const int lane = threadIdx.x & 31;
  const int t    = blockIdx.x * 8 + wid;
  const float* xr = x + (size_t)t * HID;

  float s[NE];
#pragma unroll
  for (int e = 0; e < NE; ++e) s[e] = 0.f;
  for (int k = lane; k < HID; k += 32) {
    const float xv = xr[k];
#pragma unroll
    for (int e = 0; e < NE; ++e) s[e] += xv * rw[e * HID + k];
  }
#pragma unroll
  for (int off = 16; off > 0; off >>= 1) {
#pragma unroll
    for (int e = 0; e < NE; ++e) s[e] += __shfl_xor(s[e], off, 32);
  }
  if (lane == 0) {
    float m = s[0]; int am = 0;
#pragma unroll
    for (int e = 1; e < NE; ++e) if (s[e] > m) { m = s[e]; am = e; }
    float p[NE]; float sum = 0.f;
#pragma unroll
    for (int e = 0; e < NE; ++e) { p[e] = expf(s[e] - m); sum += p[e]; }
    const float inv = 1.f / sum;
#pragma unroll
    for (int e = 0; e < NE; ++e) probs[(size_t)t * NE + e] = p[e] * inv;
    gate[t] = p[am] * inv;
    eidx[t] = am;
  }
}

// ---------------------------------------------------------------------------
// Kernel 2: scan (single block, deterministic).
// meta[0..7]=counts, meta[8..15]=segment offsets, meta[16..23]=cursors
// ---------------------------------------------------------------------------
__global__ __launch_bounds__(256) void scan_kernel(
    const float* __restrict__ probs, const int* __restrict__ eidx,
    int* __restrict__ meta, int* __restrict__ tile_expert,
    int* __restrict__ row_token, float* __restrict__ aux)
{
  __shared__ int   sc[NE];
  __shared__ float ssum[NE];
  const int tid = threadIdx.x;
  if (tid < NE) sc[tid] = 0;
  __syncthreads();

  for (int t = tid; t < T_TOK; t += 256) atomicAdd(&sc[eidx[t]], 1);
  if (tid < NE) {          // fixed-order column sums -> deterministic aux loss
    float s = 0.f;
    for (int t = 0; t < T_TOK; ++t) s += probs[(size_t)t * NE + tid];
    ssum[tid] = s;
  }
  for (int r = tid; r < TPAD; r += 256) row_token[r] = -1;
  __syncthreads();

  if (tid == 0) {
    int off = 0;
    float loss = 0.f;
    for (int e = 0; e < NE; ++e) {
      const int c = sc[e];
      meta[e] = c; meta[8 + e] = off; meta[16 + e] = off;
      const int pc = (c + (PADR - 1)) & ~(PADR - 1);
      for (int i = off / PADR; i < (off + pc) / PADR; ++i) tile_expert[i] = e;
      off += pc;
      loss += (ssum[e] / (float)T_TOK) * ((float)c / (float)T_TOK);
    }
    for (int i = off / PADR; i < MAX_MT; ++i) tile_expert[i] = -1;
    aux[0] = 0.01f * (float)NE * loss;
  }
}

// ---------------------------------------------------------------------------
// Kernel 3: gather token rows into expert segments, fp32 -> bf16 (vectorized).
// ---------------------------------------------------------------------------
__global__ __launch_bounds__(128) void gather_kernel(
    const float* __restrict__ x, const int* __restrict__ eidx,
    int* __restrict__ cursors, int* __restrict__ row_token, bf16* __restrict__ Xg)
{
  const int t = blockIdx.x;
  __shared__ int spos;
  if (threadIdx.x == 0) {
    const int e = eidx[t];
    const int pos = atomicAdd(&cursors[e], 1);
    row_token[pos] = t;
    spos = pos;
  }
  __syncthreads();
  const int pos = spos;
  const float4* src = (const float4*)(x + (size_t)t * HID);
  uint2* dst = (uint2*)(Xg + (size_t)pos * HID);
  for (int k = threadIdx.x; k < HID / 4; k += 128) {
    const float4 w = src[k];
    union { bf16 h[4]; uint2 v; } u;
    u.h[0] = (bf16)w.x; u.h[1] = (bf16)w.y; u.h[2] = (bf16)w.z; u.h[3] = (bf16)w.w;
    dst[k] = u.v;
  }
}

// ---------------------------------------------------------------------------
// Kernel 4: zero pad rows of Xg.
// ---------------------------------------------------------------------------
__global__ __launch_bounds__(64) void padzero_kernel(
    const int* __restrict__ row_token, bf16* __restrict__ Xg)
{
  const int r = blockIdx.x;
  if (row_token[r] != -1) return;
  uint4 z; z.x = z.y = z.z = z.w = 0u;
  uint4* p = (uint4*)(Xg + (size_t)r * HID);
  for (int i = threadIdx.x; i < HID / 8; i += 64) p[i] = z;
}

// ---------------------------------------------------------------------------
// Double-buffered WMMA GEMM. Block 256 (8 waves). Tile 64(M) x 256(N), K=32.
// Wave w: wm = w&1 (M half, 32 rows), wn = w>>1 (N quarter, 64 cols);
// 8 accumulators (32x64 per wave). A: bf16 [TPAD][KDIM]; B: bf16
// [E][ntot][KDIM]. Both staged to LDS by global_load_async_to_lds_b128
// (5 issues/wave/stage), s_wait_asynccnt 0x5 / 0x0.
// ---------------------------------------------------------------------------
template <int KDIM, bool FC2>
__global__ __launch_bounds__(256) void moe_gemm_kernel(
    const bf16* __restrict__ A,
    const bf16* __restrict__ B,
    const float* __restrict__ bias,
    const int*   __restrict__ tile_expert,
    bf16*        __restrict__ Hout,
    const int*   __restrict__ row_token,
    const float* __restrict__ gate,
    float*       __restrict__ out,
    int ntot)
{
  const int mt = blockIdx.x;
  const int e  = tile_expert[mt];
  if (e < 0) return;
  const int nbase = blockIdx.y * 256;
  const int tid  = threadIdx.x;
  const int wid  = tid >> 5;
  const int lane = tid & 31;
  const int wm   = wid & 1;        // M half (0/1)
  const int wn   = wid >> 1;       // N quarter (0..3)

  // double-buffered LDS tiles, row stride 40 bf16 (80B, 16B-aligned rows)
  __shared__ bf16 As[2][64][40];     // 10 KB
  __shared__ bf16 Bs[2][256][40];    // 40 KB
  v8f acc[2][4];
#pragma unroll
  for (int a = 0; a < 2; ++a)
#pragma unroll
    for (int b = 0; b < 4; ++b)
#pragma unroll
      for (int i = 0; i < 8; ++i) acc[a][b][i] = 0.f;

  const int l15   = lane & 15;
  const int kbase = (lane >> 4) * 8;           // 16-bit A/B fragment layout

  const unsigned long long abase = (unsigned long long)(uintptr_t)A;
  const unsigned long long bbase =
      (unsigned long long)(uintptr_t)(B + (size_t)e * ntot * KDIM);

  // staging maps: A 64x32 bf16 (256 x16B chunks, 1/thread),
  //               B 256x32 bf16 (1024 x16B chunks, 4/thread)
  const int sr = tid >> 2, sq = tid & 3;
  const uint32_t a_go = (uint32_t)((((size_t)(mt * 64 + sr)) * KDIM + sq * 8) * 2);

  auto stage = [&](int kk, int p) {
    async_g2l_b128(lds_addr32(&As[p][sr][sq * 8]),
                   a_go + (uint32_t)(kk * 2), abase);
#pragma unroll
    for (int i = 0; i < 4; ++i) {
      const int r = sr + i * 64;
      const uint32_t go =
          (uint32_t)((((size_t)(nbase + r)) * KDIM + kk + sq * 8) * 2);
      async_g2l_b128(lds_addr32(&Bs[p][r][sq * 8]), go, bbase);
    }
  };

  union frag_u { uint4 q[2]; v16bf v; };

  stage(0, 0);
  int p = 0;
  for (int kk = 0; kk < KDIM; kk += 32) {
    if (kk + 32 < KDIM) { stage(kk + 32, p ^ 1); wait_async_le5(); }
    else                { wait_async_0(); }
    __syncthreads();

    frag_u fa[2], fb[4];
#pragma unroll
    for (int i = 0; i < 2; ++i) {
      const int arow = wm * 32 + i * 16 + l15;
      fa[i].q[0] = *(const uint4*)&As[p][arow][kbase];
      fa[i].q[1] = *(const uint4*)&As[p][arow][kbase + 16];
    }
#pragma unroll
    for (int b = 0; b < 4; ++b) {
      const int brow = wn * 64 + b * 16 + l15;
      fb[b].q[0] = *(const uint4*)&Bs[p][brow][kbase];
      fb[b].q[1] = *(const uint4*)&Bs[p][brow][kbase + 16];
    }
#pragma unroll
    for (int a = 0; a < 2; ++a)
#pragma unroll
      for (int b = 0; b < 4; ++b)
        acc[a][b] = __builtin_amdgcn_wmma_f32_16x16x32_bf16(
            false, fa[a].v, false, fb[b].v, (short)0, acc[a][b], false, false);

    __syncthreads();
    p ^= 1;
  }

  // C/D layout: elem i -> m = i + 8*(lane>>4), n = lane&15
  const int m00 = mt * 64 + wm * 32 + 8 * (lane >> 4);
  const int c00 = nbase + wn * 64 + l15;
  const float* brow = bias + (size_t)e * ntot;

#pragma unroll
  for (int a = 0; a < 2; ++a) {
#pragma unroll
    for (int b = 0; b < 4; ++b) {
      const int cb = c00 + b * 16;
      const float bv = brow[cb];
#pragma unroll
      for (int i = 0; i < 8; ++i) {
        const int row = m00 + a * 16 + i;
        if constexpr (!FC2) {
          float v = acc[a][b][i] + bv;
          v = 0.5f * v * (1.0f + erff(v * 0.70710678118654752f));
          Hout[(size_t)row * FFN_ + cb] = (bf16)v;
        } else {
          const int tok = row_token[row];
          if (tok >= 0)
            out[(size_t)tok * HID + cb] = (acc[a][b][i] + bv) * gate[tok];
        }
      }
    }
  }
}

// ---------------------------------------------------------------------------
extern "C" void kernel_launch(void* const* d_in, const int* in_sizes, int n_in,
                              void* d_out, int out_size, void* d_ws, size_t ws_size,
                              hipStream_t stream)
{
  (void)in_sizes; (void)n_in; (void)out_size; (void)ws_size;

  const float* x  = (const float*)d_in[0];
  const float* rw = (const float*)d_in[1];
  const float* w1 = (const float*)d_in[2];
  const float* b1 = (const float*)d_in[3];
  const float* w2 = (const float*)d_in[4];
  const float* b2 = (const float*)d_in[5];
  float* out = (float*)d_out;

  char* ws = (char*)d_ws;
  float* probs   = (float*)(ws + WS_PROBS);
  float* gate    = (float*)(ws + WS_GATE);
  int*   eidx    = (int*)  (ws + WS_EIDX);
  int*   meta    = (int*)  (ws + WS_META);
  int*   tileExp = (int*)  (ws + WS_TILE);
  int*   rowTok  = (int*)  (ws + WS_ROWTK);
  bf16*  Xg      = (bf16*) (ws + WS_XG);
  bf16*  Hbuf    = (bf16*) (ws + WS_H);
  bf16*  w1b     = (bf16*) (ws + WS_W1B);
  bf16*  w2b     = (bf16*) (ws + WS_W2B);

  float* aux = out + (size_t)T_TOK * HID;

  // weight conversion: bf16 copies (128MB total) -> live in 192MB L2
  cvt_kernel<<<2048, 256, 0, stream>>>(w1, w1b, NE * FFN_ * HID / 4);
  cvt_kernel<<<2048, 256, 0, stream>>>(w2, w2b, NE * HID * FFN_ / 4);

  router_kernel<<<T_TOK / 8, 256, 0, stream>>>(x, rw, probs, gate, eidx);
  scan_kernel<<<1, 256, 0, stream>>>(probs, eidx, meta, tileExp, rowTok, aux);
  gather_kernel<<<T_TOK, 128, 0, stream>>>(x, eidx, meta + 16, rowTok, Xg);
  padzero_kernel<<<TPAD, 64, 0, stream>>>(rowTok, Xg);

  dim3 g1(MAX_MT, FFN_ / 256);
  moe_gemm_kernel<HID, false><<<g1, 256, 0, stream>>>(
      Xg, w1b, b1, tileExp, Hbuf, nullptr, nullptr, nullptr, FFN_);

  dim3 g2(MAX_MT, HID / 256);
  moe_gemm_kernel<FFN_, true><<<g2, 256, 0, stream>>>(
      Hbuf, w2b, b2, tileExp, nullptr, rowTok, gate, out, HID);
}
